// EncoderDecoder_7619271983179
// MI455X (gfx1250) — matrix operand order repaired and checked by reading, not compiled, not run on hardware
//
#include <hip/hip_runtime.h>
#include <hip/hip_bf16.h>

// ---------------------------------------------------------------------------
// Seq2seq LSTM encoder/decoder for MI455X (gfx1250, wave32, WMMA bf16).
//   B=256, T=64, H=1024, V=128, L=32
// Strategy:
//   * fold x@W+b into per-vocab table (emb@W+b): (128 x 4096) f32, one-time
//   * recurrent h@U done with v_wmma_f32_16x16x32_bf16; U pre-packed to
//     B-fragment order in bf16 (L2-resident, 8MB)
//   * h kept in A-fragment-ordered bf16 (double buffered), staged via LDS
//   * each wave computes same 16-col slice of all 4 gates -> gates local
//   * K-loop software-pipelined: next-iteration A/B fragments prefetched
//     into registers so WMMAs overlap L2 latency (partial s_wait_loadcnt)
// ---------------------------------------------------------------------------

typedef __attribute__((ext_vector_type(16))) __bf16 v16bf;
typedef __attribute__((ext_vector_type(8)))  float  v8f;

#define Bsz   256
#define Tlen  64
#define Hdim  1024
#define Vocab 128
#define Ldec  32
#define G4H   4096
#define KT    32    /* Hdim / 32 k-tiles   */
#define MT    16    /* Bsz / 16 m-tiles    */

__device__ __forceinline__ unsigned short f2bf(float f) {
  unsigned u = __float_as_uint(f);
  u += 0x7FFFu + ((u >> 16) & 1u);           // round-to-nearest-even
  return (unsigned short)(u >> 16);
}
__device__ __forceinline__ float sigmoid_f(float x) {
  return 1.0f / (1.0f + __expf(-x));
}
__device__ __forceinline__ float tanh_f(float x) {
  x = fminf(fmaxf(x, -15.0f), 15.0f);
  float e = __expf(2.0f * x);
  return (e - 1.0f) / (e + 1.0f);
}
#define WMMA_BF16(A, Bm, C) \
  __builtin_amdgcn_wmma_f32_16x16x32_bf16(false, (A), false, (Bm), (short)0, (C), false, false)

// ---------------------------------------------------------------------------
// table[v][n] = sum_k emb[v][k] * W[k][n] + bias[n]      (128 x 4096, f32)
// ---------------------------------------------------------------------------
__global__ void __launch_bounds__(256) prep_table_kernel(
    const float* __restrict__ emb, const float* __restrict__ W,
    const float* __restrict__ bias, float* __restrict__ table) {
  int tid = blockIdx.x * blockDim.x + threadIdx.x;   // < 128*4096
  int n = tid & (G4H - 1);
  int v = tid >> 12;
  const float* er = emb + (size_t)v * Hdim;
  float acc = bias[n];
#pragma unroll 8
  for (int k = 0; k < Hdim; ++k) acc = fmaf(er[k], W[(size_t)k * G4H + n], acc);
  table[tid] = acc;
}

// ---------------------------------------------------------------------------
// Pack f32 row-major W (Hdim x N) into bf16 WMMA B-fragment order:
//   Wpk[((nt*KT + kt)*32 + lane)*16 + e]  <-  W[k][col]
//   lane 0..15 : K = kt*32 + e      , N = nt*16 + lane
//   lane 16..31: K = kt*32 + 16 + e , N = nt*16 + lane-16
// ---------------------------------------------------------------------------
__global__ void __launch_bounds__(256) pack_b_kernel(
    const float* __restrict__ W, unsigned short* __restrict__ Wpk, int N) {
  int tid  = blockIdx.x * blockDim.x + threadIdx.x;  // < Hdim*N
  int e    = tid & 15;
  int n    = (tid >> 4) & 31;
  int rest = tid >> 9;
  int kt   = rest & 31;
  int nt   = rest >> 5;
  int k    = kt * 32 + ((n >> 4) << 4) + e;
  int col  = nt * 16 + (n & 15);
  Wpk[tid] = f2bf(W[(size_t)k * N + col]);
}

// ---------------------------------------------------------------------------
// Zero-init c, hA buffer 0; tok = delimiter (V-1)
// ---------------------------------------------------------------------------
__global__ void __launch_bounds__(256) init_kernel(
    float* __restrict__ c, unsigned short* __restrict__ hA0, int* __restrict__ tok) {
  int tid = blockIdx.x * blockDim.x + threadIdx.x;   // 256*1024 threads
  c[tid] = 0.0f;
  hA0[tid] = 0;                                      // bf16 0.0
  if (tid < Bsz) tok[tid] = Vocab - 1;
}

// ---------------------------------------------------------------------------
// One LSTM step:  z = table[idx[b]] + h @ U ; gates ; masked h/c update.
// grid = 128 blocks (16 mtiles x 8 jgroups), 256 threads (8 waves).
// Wave handles jtile = jgroup*8+wave: 16 hidden columns across ALL 4 gates.
// A (h, bf16, fragment-ordered) staged in LDS, shared by 8 waves.
// ---------------------------------------------------------------------------
__global__ void __launch_bounds__(256) lstm_step_kernel(
    const unsigned short* __restrict__ hA_in, unsigned short* __restrict__ hA_out,
    float* __restrict__ c, const float* __restrict__ table,
    const unsigned short* __restrict__ Upk,
    const int* __restrict__ idx, int idx_stride, int idx_off) {
  __shared__ __align__(16) unsigned short As[16 * Hdim];   // 32 KB
  const int tid    = threadIdx.x;
  const int lane   = tid & 31;
  const int wave   = tid >> 5;
  const int mtile  = blockIdx.x >> 3;
  const int jgroup = blockIdx.x & 7;
  const int jtile  = jgroup * 8 + wave;

  // cooperative LDS stage of this mtile's A fragments (contiguous 32 KB)
  {
    const uint4* src = (const uint4*)(hA_in + (size_t)mtile * 16 * Hdim);
    uint4* dst = (uint4*)As;
#pragma unroll
    for (int i = 0; i < 8; ++i) dst[tid + i * 256] = src[tid + i * 256];
  }
  __syncthreads();

  const unsigned short* lA  = As + lane * 16;
  const size_t lb = (size_t)lane * 16;
  const unsigned short* pB0 = Upk + (size_t)((0 * 64 + jtile) * KT) * 512 + lb;
  const unsigned short* pB1 = Upk + (size_t)((1 * 64 + jtile) * KT) * 512 + lb;
  const unsigned short* pB2 = Upk + (size_t)((2 * 64 + jtile) * KT) * 512 + lb;
  const unsigned short* pB3 = Upk + (size_t)((3 * 64 + jtile) * KT) * 512 + lb;

  v8f acc0 = {0.f,0.f,0.f,0.f,0.f,0.f,0.f,0.f};
  v8f acc1 = acc0, acc2 = acc0, acc3 = acc0;

  // software-pipelined K loop: prefetch kt+1 fragments before consuming kt
  v16bf a  = *(const v16bf*)(lA);
  v16bf b0 = *(const v16bf*)(pB0);
  v16bf b1 = *(const v16bf*)(pB1);
  v16bf b2 = *(const v16bf*)(pB2);
  v16bf b3 = *(const v16bf*)(pB3);
#pragma unroll 1
  for (int kt = 0; kt < KT - 1; ++kt) {
    const size_t nb = (size_t)(kt + 1) * 512;
    v16bf an  = *(const v16bf*)(lA + nb);
    v16bf bn0 = *(const v16bf*)(pB0 + nb);
    v16bf bn1 = *(const v16bf*)(pB1 + nb);
    v16bf bn2 = *(const v16bf*)(pB2 + nb);
    v16bf bn3 = *(const v16bf*)(pB3 + nb);
    acc0 = WMMA_BF16(a, b0, acc0);
    acc1 = WMMA_BF16(a, b1, acc1);
    acc2 = WMMA_BF16(a, b2, acc2);
    acc3 = WMMA_BF16(a, b3, acc3);
    a = an; b0 = bn0; b1 = bn1; b2 = bn2; b3 = bn3;
  }
  acc0 = WMMA_BF16(a, b0, acc0);
  acc1 = WMMA_BF16(a, b1, acc1);
  acc2 = WMMA_BF16(a, b2, acc2);
  acc3 = WMMA_BF16(a, b3, acc3);

  // gate phase: lane holds (M = r + 8*(lane>>4), N = lane&15) of each acc
  const int laneHi = lane >> 4;
  const int nl     = lane & 15;
  const int j      = jtile * 16 + nl;          // hidden index
  const int ktile  = j >> 5;
  const int kl     = j & 31;
  const int half   = kl >> 4;
  const int e      = kl & 15;

#pragma unroll
  for (int r = 0; r < 8; ++r) {
    int bl   = r + (laneHi << 3);              // row within mtile
    int brow = mtile * 16 + bl;                // batch row
    int vrow = idx[brow * idx_stride + idx_off];
    const float* trow = table + (size_t)vrow * G4H + j;
    size_t cix  = (size_t)brow * Hdim + j;
    size_t aoff = ((size_t)(mtile * KT + ktile) * 32 + bl + (half << 4)) * 16 + e;
    if (vrow != 0) {
      float iv = sigmoid_f(acc0[r] + trow[0]);
      float fv = sigmoid_f(acc1[r] + trow[1024]);
      float gv = tanh_f   (acc2[r] + trow[2048]);
      float ov = sigmoid_f(acc3[r] + trow[3072]);
      float c2 = fmaf(fv, c[cix], iv * gv);
      float h2 = ov * tanh_f(c2);
      c[cix] = c2;
      hA_out[aoff] = f2bf(h2);
    } else {
      hA_out[aoff] = hA_in[aoff];              // masked: carry state
    }
  }
}

// ---------------------------------------------------------------------------
// Decoder output head: logits = h2 @ out_W + out_b ; softmax ; argmax -> tok.
// grid = 16 blocks (one per mtile), 8 waves = 8 ntiles of V=128.
// ---------------------------------------------------------------------------
__global__ void __launch_bounds__(256) dec_out_kernel(
    const unsigned short* __restrict__ hA, const unsigned short* __restrict__ Wpk,
    const float* __restrict__ out_b, int* __restrict__ tok,
    float* __restrict__ out, int step) {
  __shared__ __align__(16) unsigned short As[16 * Hdim];   // 32 KB
  __shared__ float logits[16 * Vocab];                     // 8 KB
  const int tid   = threadIdx.x;
  const int lane  = tid & 31;
  const int wave  = tid >> 5;     // ntile 0..7
  const int mtile = blockIdx.x;

  {
    const uint4* src = (const uint4*)(hA + (size_t)mtile * 16 * Hdim);
    uint4* dst = (uint4*)As;
#pragma unroll
    for (int i = 0; i < 8; ++i) dst[tid + i * 256] = src[tid + i * 256];
  }
  __syncthreads();

  const unsigned short* lA = As + lane * 16;
  const unsigned short* pB = Wpk + (size_t)(wave * KT) * 512 + (size_t)lane * 16;

  v8f acc = {0.f,0.f,0.f,0.f,0.f,0.f,0.f,0.f};
  v16bf a = *(const v16bf*)(lA);
  v16bf b = *(const v16bf*)(pB);
#pragma unroll 1
  for (int kt = 0; kt < KT - 1; ++kt) {
    const size_t nb = (size_t)(kt + 1) * 512;
    v16bf an = *(const v16bf*)(lA + nb);
    v16bf bn = *(const v16bf*)(pB + nb);
    acc = WMMA_BF16(a, b, acc);
    a = an; b = bn;
  }
  acc = WMMA_BF16(a, b, acc);

  const int laneHi = lane >> 4;
  const int n = wave * 16 + (lane & 15);
  float bn = out_b[n];
#pragma unroll
  for (int r = 0; r < 8; ++r)
    logits[(r + (laneHi << 3)) * Vocab + n] = acc[r] + bn;
  __syncthreads();

  if (tid < 16) {
    int brow = mtile * 16 + tid;
    const float* lr = logits + tid * Vocab;
    float mx = lr[0]; int best = 0;
    for (int v = 1; v < Vocab; ++v) { float x = lr[v]; if (x > mx) { mx = x; best = v; } }
    float s = 0.0f;
    for (int v = 0; v < Vocab; ++v) s += __expf(lr[v] - mx);
    float inv = 1.0f / s;
    float* yr = out + (size_t)brow * (Ldec * Vocab) + (size_t)step * Vocab;
    for (int v = 0; v < Vocab; ++v) yr[v] = __expf(lr[v] - mx) * inv;
    tok[brow] = best;
    out[(size_t)Bsz * Ldec * Vocab + (size_t)step * Bsz + brow] = (float)best;
  }
}

// ---------------------------------------------------------------------------
extern "C" void kernel_launch(void* const* d_in, const int* in_sizes, int n_in,
                              void* d_out, int out_size, void* d_ws, size_t ws_size,
                              hipStream_t stream) {
  (void)in_sizes; (void)n_in; (void)out_size; (void)ws_size;
  const int*   inputs  = (const int*)  d_in[0];
  const float* enc_emb = (const float*)d_in[2];
  const float* enc_W   = (const float*)d_in[3];
  const float* enc_U   = (const float*)d_in[4];
  const float* enc_b   = (const float*)d_in[5];
  const float* dec_emb = (const float*)d_in[6];
  const float* dec_W   = (const float*)d_in[7];
  const float* dec_U   = (const float*)d_in[8];
  const float* dec_b   = (const float*)d_in[9];
  const float* out_W   = (const float*)d_in[10];
  const float* out_bv  = (const float*)d_in[11];
  float* out = (float*)d_out;

  char* ws = (char*)d_ws;
  size_t off = 0;
  auto alloc = [&](size_t bytes) -> char* {
    char* p = ws + off;
    off += (bytes + 255) & ~(size_t)255;
    return p;
  };
  float*          encT = (float*)         alloc((size_t)Vocab * G4H * 4);
  float*          decT = (float*)         alloc((size_t)Vocab * G4H * 4);
  unsigned short* Uenc = (unsigned short*)alloc((size_t)Hdim * G4H * 2);
  unsigned short* Udec = (unsigned short*)alloc((size_t)Hdim * G4H * 2);
  unsigned short* Wout = (unsigned short*)alloc((size_t)Hdim * Vocab * 2);
  float*          cbuf = (float*)         alloc((size_t)Bsz * Hdim * 4);
  unsigned short* hA0  = (unsigned short*)alloc((size_t)Bsz * Hdim * 2);
  unsigned short* hA1  = (unsigned short*)alloc((size_t)Bsz * Hdim * 2);
  int*            tok  = (int*)           alloc((size_t)Bsz * 4);

  // one-time prep (recomputed every call: deterministic, cheap)
  prep_table_kernel<<<(Vocab * G4H) / 256, 256, 0, stream>>>(enc_emb, enc_W, enc_b, encT);
  prep_table_kernel<<<(Vocab * G4H) / 256, 256, 0, stream>>>(dec_emb, dec_W, dec_b, decT);
  pack_b_kernel<<<(Hdim * G4H) / 256, 256, 0, stream>>>(enc_U, Uenc, G4H);
  pack_b_kernel<<<(Hdim * G4H) / 256, 256, 0, stream>>>(dec_U, Udec, G4H);
  pack_b_kernel<<<(Hdim * Vocab) / 256, 256, 0, stream>>>(out_W, Wout, Vocab);
  init_kernel<<<(Bsz * Hdim) / 256, 256, 0, stream>>>(cbuf, hA0, tok);

  unsigned short* hin  = hA0;
  unsigned short* hout = hA1;

  // encoder: 64 recurrent steps
  for (int t = 0; t < Tlen; ++t) {
    lstm_step_kernel<<<128, 256, 0, stream>>>(hin, hout, cbuf, encT, Uenc,
                                              inputs, Tlen, t);
    unsigned short* tmp = hin; hin = hout; hout = tmp;
  }

  // decoder: 32 recurrent steps with greedy token feedback
  for (int s = 0; s < Ldec; ++s) {
    lstm_step_kernel<<<128, 256, 0, stream>>>(hin, hout, cbuf, decT, Udec,
                                              tok, 1, 0);
    dec_out_kernel<<<16, 256, 0, stream>>>(hout, Wout, out_bv, tok, out, s);
    unsigned short* tmp = hin; hin = hout; hout = tmp;
  }
}
